// GAT_60335700574379
// MI455X (gfx1250) — compile-verified, matrix operand order checked by reading, not run on hardware
//
#include <hip/hip_runtime.h>
#include <hip/hip_bf16.h>
#include <cmath>
#include <stdint.h>

#define THREADS 256
#define NEG_SLOPE 0.2f
#define STRIP 8   // M-tiles per wave in the GEMM (B fragments held in registers)

typedef __attribute__((ext_vector_type(16))) _Float16 v16h;
typedef __attribute__((ext_vector_type(8)))  _Float16 v8h;
typedef __attribute__((ext_vector_type(8)))  float    v8f;

union Frag16 { v16h v; v8h h[2]; };

// native f32 atomic add (global_atomic_add_f32); accumulators are coarse-grained ws memory
__device__ __forceinline__ void atomAddF(float* a, float v) { unsafeAtomicAdd(a, v); }

// ---------------- utility kernels ----------------
__global__ void fill_f32(float* __restrict__ p, long long n, float v) {
  long long i = (long long)blockIdx.x * blockDim.x + threadIdx.x;
  if (i < n) p[i] = v;
}

__global__ void recip_inplace(float* __restrict__ p, long long n) {
  long long i = (long long)blockIdx.x * blockDim.x + threadIdx.x;
  if (i < n) p[i] = 1.f / (p[i] + 1e-16f);
}

__global__ void cast_f32_to_f16(const float* __restrict__ s, _Float16* __restrict__ d, long long n) {
  long long i = (long long)blockIdx.x * blockDim.x + threadIdx.x;
  if (i < n) d[i] = (_Float16)s[i];
}

// Wt[n*K + k] = W[k*cols + n]  (i runs over cols*K; writes coalesced)
__global__ void transpose_cast_f16(const float* __restrict__ W, _Float16* __restrict__ Wt,
                                   int K, int cols) {
  int i = blockIdx.x * blockDim.x + threadIdx.x;
  if (i >= K * cols) return;
  int n = i / K, k = i - n * K;
  Wt[i] = (_Float16)W[(size_t)k * cols + n];
}

// ---------------- WMMA GEMM: C[M x Nout] = A[M x 128] @ Bt[Nout x 128]^T ----------------
// One wave pins the 4 K-step B fragments for its 16-wide output column in registers,
// then sweeps STRIP consecutive 16x16 M-tiles. All control flow is wave-uniform so
// EXEC stays all-ones for every v_wmma (ISA requirement).
__global__ void __launch_bounds__(THREADS) gemm_wmma(const _Float16* __restrict__ A,
                                                     const _Float16* __restrict__ Bt,
                                                     float* __restrict__ C,
                                                     int nTilesN, int ldc,
                                                     int mTiles, int strips) {
  int wave = (int)((blockIdx.x * blockDim.x + threadIdx.x) >> 5);
  int lane = threadIdx.x & 31;
  int stripId = wave / nTilesN;
  if (stripId >= strips) return;                       // wave-uniform
  int tn = (wave - stripId * nTilesN) << 4;
  int l15 = lane & 15;
  int hi  = (lane >> 4) & 1;
  // ISA 16-bit fragment layout: lanes 0-15 hold K {0..7, 16..23}; lanes 16-31 hold K {8..15, 24..31}
  const _Float16* brow = Bt + (size_t)(tn + l15) * 128 + hi * 8;
  Frag16 b[4];
#pragma unroll
  for (int kk = 0; kk < 4; ++kk) {
    b[kk].h[0] = *(const v8h*)(brow + kk * 32);
    b[kk].h[1] = *(const v8h*)(brow + kk * 32 + 16);
  }
  int tm0 = stripId * STRIP;
#pragma unroll 1
  for (int t = 0; t < STRIP; ++t) {
    int tmTile = tm0 + t;
    if (tmTile >= mTiles) break;                       // wave-uniform
    int tm = tmTile << 4;
    const _Float16* arow = A + (size_t)(tm + l15) * 128 + hi * 8;
    v8f acc = {};
#pragma unroll
    for (int kk = 0; kk < 4; ++kk) {
      Frag16 a;
      a.h[0] = *(const v8h*)(arow + kk * 32);
      a.h[1] = *(const v8h*)(arow + kk * 32 + 16);
      acc = __builtin_amdgcn_wmma_f32_16x16x32_f16(false, a.v, false, b[kk].v,
                                                   (short)0, acc, false, false);
    }
    // C/D layout: VGPR g -> row tm + g + hi*8, col tn + (lane&15)
    float* crow = C + (size_t)(tm + hi * 8) * ldc + tn + l15;
#pragma unroll
    for (int g = 0; g < 8; ++g) crow[(size_t)g * ldc] = acc[g];
  }
}

// ---------------- attention coefficients: a_s[n,h] = <h[n,h,:], att_s[h,:]> ----------------
template <int H, int C>
__global__ void attn_coef(const float* __restrict__ h,
                          const float* __restrict__ att_s,
                          const float* __restrict__ att_d,
                          float* __restrict__ as_, float* __restrict__ ad_, int n) {
  int i = blockIdx.x * blockDim.x + threadIdx.x;
  if (i >= n * H) return;
  int node = i / H, hd = i - node * H;
  const float* hp = h + (size_t)node * (H * C) + hd * C;
  const float* sp = att_s + hd * C;
  const float* dp = att_d + hd * C;
  float s = 0.f, d = 0.f;
#pragma unroll
  for (int c = 0; c < C; ++c) { float v = hp[c]; s += v * sp[c]; d += v * dp[c]; }
  as_[i] = s; ad_[i] = d;
}

// ---------------- edge helpers ----------------
__device__ __forceinline__ void edge_sd(const long long* __restrict__ ei, long long E,
                                        long long e, int& s, int& d) {
  if (e < E) { s = (int)ei[e]; d = (int)ei[E + e]; }
  else       { int nn = (int)(e - E); s = nn; d = nn; }  // self-loops appended
}
__device__ __forceinline__ float lrelu(float x) { return x > 0.f ? x : NEG_SLOPE * x; }
__device__ __forceinline__ void atomicMaxF(float* a, float v) {
  // monotonic int-key trick; correct with -inf init
  if (v >= 0.f) atomicMax((int*)a, __float_as_int(v));
  else          atomicMin((unsigned int*)a, __float_as_uint(v));
}

template <int H>
__global__ void edge_max(const long long* __restrict__ ei, long long E, int N,
                         const float* __restrict__ as_, const float* __restrict__ ad_,
                         float* __restrict__ m) {
  long long tid = (long long)blockIdx.x * blockDim.x + threadIdx.x;
  long long tot = (E + N) * H;
  if (tid >= tot) return;
  long long e = tid / H; int hd = (int)(tid - e * H);
  int s, d; edge_sd(ei, E, e, s, d);
  float sc = lrelu(as_[(size_t)s * H + hd] + ad_[(size_t)d * H + hd]);
  atomicMaxF(&m[(size_t)d * H + hd], sc);
}

// p = exp(sc - m[dst]) computed ONCE per (edge, head): stored for the aggregation pass
// and accumulated into the softmax denominator.
template <int H>
__global__ void edge_alpha(const long long* __restrict__ ei, long long E, int N,
                           const float* __restrict__ as_, const float* __restrict__ ad_,
                           const float* __restrict__ m, float* __restrict__ den,
                           float* __restrict__ pbuf) {
  long long tid = (long long)blockIdx.x * blockDim.x + threadIdx.x;
  long long tot = (E + N) * H;
  if (tid >= tot) return;
  long long e = tid / H; int hd = (int)(tid - e * H);
  int s, d; edge_sd(ei, E, e, s, d);
  float sc = lrelu(as_[(size_t)s * H + hd] + ad_[(size_t)d * H + hd]);
  float p  = __expf(sc - m[(size_t)d * H + hd]);
  atomAddF(&den[(size_t)d * H + hd], p);
  pbuf[tid] = p;
}

// out[dst,f] += (p * inv_den[dst,h]) * h[src,f]; thread per (edge, feature)
template <int H, int C>
__global__ void edge_aggr(const long long* __restrict__ ei, long long E, int N,
                          const float* __restrict__ pbuf, const float* __restrict__ invden,
                          const float* __restrict__ hfeat, float* __restrict__ agg) {
  const int F = H * C;
  long long tid = (long long)blockIdx.x * blockDim.x + threadIdx.x;
  long long tot = (E + N) * F;
  if (tid >= tot) return;
  long long e = tid / F; int f = (int)(tid - e * F); int hd = f / C;
  int s, d; edge_sd(ei, E, e, s, d);
  float alpha = pbuf[e * H + hd] * invden[(size_t)d * H + hd];
  atomAddF(&agg[(size_t)d * F + f], alpha * hfeat[(size_t)s * F + f]);
}

// agg + bias -> relu -> f16 (layer-2 GEMM input); F fixed = 128
__global__ void bias_relu_cast(const float* __restrict__ agg, const float* __restrict__ bias,
                               _Float16* __restrict__ outh, long long n) {
  long long i = (long long)blockIdx.x * blockDim.x + threadIdx.x;
  if (i >= n) return;
  int f = (int)(i & 127);
  float v = agg[i] + bias[f];
  outh[i] = (_Float16)fmaxf(v, 0.f);
}

// in-place log_softmax over rows of 64, one wave32 per row (lane covers cols l and l+32)
__global__ void logsoftmax64(float* __restrict__ out, const float* __restrict__ bias, int n) {
  int wave = (int)((blockIdx.x * blockDim.x + threadIdx.x) >> 5);
  int lane = threadIdx.x & 31;
  if (wave >= n) return;
  float* row = out + (size_t)wave * 64;
  float v0 = row[lane]      + bias[lane];
  float v1 = row[lane + 32] + bias[lane + 32];
  float mx = fmaxf(v0, v1);
#pragma unroll
  for (int off = 16; off > 0; off >>= 1) mx = fmaxf(mx, __shfl_xor(mx, off, 32));
  float sm = __expf(v0 - mx) + __expf(v1 - mx);
#pragma unroll
  for (int off = 16; off > 0; off >>= 1) sm += __shfl_xor(sm, off, 32);
  float lse = mx + __logf(sm);
  row[lane]      = v0 - lse;
  row[lane + 32] = v1 - lse;
}

// ---------------- host launch ----------------
static inline int nblk(long long n, int t) { return (int)((n + t - 1) / t); }

extern "C" void kernel_launch(void* const* d_in, const int* in_sizes, int n_in,
                              void* d_out, int out_size, void* d_ws, size_t ws_size,
                              hipStream_t stream) {
  const float*     x   = (const float*)d_in[0];
  const long long* ei  = (const long long*)d_in[1];
  const float*     W1  = (const float*)d_in[2];
  const float*     as1 = (const float*)d_in[3];
  const float*     ad1 = (const float*)d_in[4];
  const float*     b1  = (const float*)d_in[5];
  const float*     W2  = (const float*)d_in[6];
  const float*     as2 = (const float*)d_in[7];
  const float*     ad2 = (const float*)d_in[8];
  const float*     b2  = (const float*)d_in[9];
  float* out = (float*)d_out;
  (void)n_in; (void)out_size; (void)ws_size;

  const int       N  = in_sizes[0] / 128;   // 100000 (divisible by 16)
  const long long E  = in_sizes[1] / 2;     // 1.6M
  const long long E2 = E + N;               // with self-loops

  char* ws = (char*)d_ws;
  size_t off = 0;
  auto alloc = [&](size_t bytes) { char* p = ws + off; off = (off + bytes + 255) & ~(size_t)255; return p; };
  _Float16* xh    = (_Float16*)alloc((size_t)N * 128 * 2);
  _Float16* w1t   = (_Float16*)alloc((size_t)128 * 128 * 2);
  _Float16* w2t   = (_Float16*)alloc((size_t)64 * 128 * 2);
  float*    h1    = (float*)alloc((size_t)N * 128 * 4);
  float*    a1s   = (float*)alloc((size_t)N * 8 * 4);
  float*    a1d   = (float*)alloc((size_t)N * 8 * 4);
  float*    m1    = (float*)alloc((size_t)N * 8 * 4);
  float*    den1  = (float*)alloc((size_t)N * 8 * 4);
  float*    agg1  = (float*)alloc((size_t)N * 128 * 4);
  _Float16* agg1h = (_Float16*)alloc((size_t)N * 128 * 2);
  float*    h2    = (float*)alloc((size_t)N * 64 * 4);
  float*    a2s   = (float*)alloc((size_t)N * 4);
  float*    a2d   = (float*)alloc((size_t)N * 4);
  float*    m2    = (float*)alloc((size_t)N * 4);
  float*    den2  = (float*)alloc((size_t)N * 4);
  float*    pbuf  = (float*)alloc((size_t)E2 * 8 * 4);  // per-(edge,head) softmax numerators (reused by layer 2)

  // re-initialize every call (graph-replay safe)
  fill_f32<<<nblk((long long)N * 8,   THREADS), THREADS, 0, stream>>>(m1,   (long long)N * 8,   -INFINITY);
  fill_f32<<<nblk((long long)N * 8,   THREADS), THREADS, 0, stream>>>(den1, (long long)N * 8,   0.f);
  fill_f32<<<nblk((long long)N * 128, THREADS), THREADS, 0, stream>>>(agg1, (long long)N * 128, 0.f);
  fill_f32<<<nblk((long long)N,       THREADS), THREADS, 0, stream>>>(m2,   (long long)N,       -INFINITY);
  fill_f32<<<nblk((long long)N,       THREADS), THREADS, 0, stream>>>(den2, (long long)N,       0.f);
  fill_f32<<<nblk((long long)N * 64,  THREADS), THREADS, 0, stream>>>(out,  (long long)N * 64,  0.f);

  // precision prep for WMMA
  cast_f32_to_f16<<<nblk((long long)N * 128, THREADS), THREADS, 0, stream>>>(x, xh, (long long)N * 128);
  transpose_cast_f16<<<nblk(128 * 128, THREADS), THREADS, 0, stream>>>(W1, w1t, 128, 128);
  transpose_cast_f16<<<nblk(64 * 128,  THREADS), THREADS, 0, stream>>>(W2, w2t, 128, 64);

  const int mTiles = N / 16;                       // 6250
  const int strips = (mTiles + STRIP - 1) / STRIP; // 782

  // ----- layer 1 -----
  {
    long long waves = (long long)strips * 8;
    gemm_wmma<<<nblk(waves * 32, THREADS), THREADS, 0, stream>>>(xh, w1t, h1, 8, 128, mTiles, strips);
  }
  attn_coef<8, 16><<<nblk((long long)N * 8, THREADS), THREADS, 0, stream>>>(h1, as1, ad1, a1s, a1d, N);
  edge_max<8>  <<<nblk(E2 * 8,   THREADS), THREADS, 0, stream>>>(ei, E, N, a1s, a1d, m1);
  edge_alpha<8><<<nblk(E2 * 8,   THREADS), THREADS, 0, stream>>>(ei, E, N, a1s, a1d, m1, den1, pbuf);
  recip_inplace<<<nblk((long long)N * 8, THREADS), THREADS, 0, stream>>>(den1, (long long)N * 8);
  edge_aggr<8, 16><<<nblk(E2 * 128, THREADS), THREADS, 0, stream>>>(ei, E, N, pbuf, den1, h1, agg1);
  bias_relu_cast<<<nblk((long long)N * 128, THREADS), THREADS, 0, stream>>>(agg1, b1, agg1h, (long long)N * 128);

  // ----- layer 2 -----
  {
    long long waves = (long long)strips * 4;
    gemm_wmma<<<nblk(waves * 32, THREADS), THREADS, 0, stream>>>(agg1h, w2t, h2, 4, 64, mTiles, strips);
  }
  attn_coef<1, 64><<<nblk((long long)N, THREADS), THREADS, 0, stream>>>(h2, as2, ad2, a2s, a2d, N);
  edge_max<1>  <<<nblk(E2,      THREADS), THREADS, 0, stream>>>(ei, E, N, a2s, a2d, m2);
  edge_alpha<1><<<nblk(E2,      THREADS), THREADS, 0, stream>>>(ei, E, N, a2s, a2d, m2, den2, pbuf);
  recip_inplace<<<nblk((long long)N, THREADS), THREADS, 0, stream>>>(den2, (long long)N);
  edge_aggr<1, 64><<<nblk(E2 * 64, THREADS), THREADS, 0, stream>>>(ei, E, N, pbuf, den2, h2, out);

  logsoftmax64<<<nblk((long long)N * 32, THREADS), THREADS, 0, stream>>>(out, b2, N);
}
